// MPNNLSTM_46368466927681
// MI455X (gfx1250) — compile-verified, matrix operand order checked by reading, not run on hardware
//
#include <hip/hip_runtime.h>
#include <hip/hip_bf16.h>
#include <math.h>

#define NN 10000          // nodes
#define EE 160000         // edges
#define EPN (EE + NN)     // edges + self loops
#define TT 8
#define BB 2
#define NH 64
#define GROWS (BB * TT * NN)   // 160000 bt-major rows for GCN stages
#define BNR   (BB * NN)        // 20000 rows for LSTM/FC

typedef __attribute__((ext_vector_type(16))) _Float16 v16h;
typedef __attribute__((ext_vector_type(8)))  _Float16 v8h;
typedef __attribute__((ext_vector_type(8)))  float    v8f;

__device__ inline v8f zero8() {
  v8f z;
#pragma unroll
  for (int i = 0; i < 8; ++i) z[i] = 0.f;
  return z;
}

__device__ inline float sigm(float x) { return 1.f / (1.f + __expf(-x)); }

// 16x32 f16 fragment from a row-major source (stride ld halves). Lane holds
// row (lane&15); halves 0-7 = K k0..k0+7, halves 8-15 = K k0+16..k0+23 with
// k0 = (lane>>4)*8. Both runs are contiguous -> two 16-byte vector loads.
// All call sites guarantee 16-byte alignment of p + row*ld + k0.
__device__ inline v16h frag_rows(const _Float16* p, int ld) {
  const int lane = threadIdx.x & 31;
  const _Float16* q = p + (lane & 15) * ld + ((lane >> 4) << 3);
  v8h lo = *(const v8h*)(q);
  v8h hi = *(const v8h*)(q + 16);
  return __builtin_shufflevector(lo, hi, 0, 1, 2, 3, 4, 5, 6, 7,
                                 8, 9, 10, 11, 12, 13, 14, 15);
}

// ------------------------------------------------------------------ utility
__global__ void __launch_bounds__(256) k_zero(float* p, long long n) {
  long long i = (long long)blockIdx.x * blockDim.x + threadIdx.x;
  if (i < n) p[i] = 0.f;
}

__global__ void __launch_bounds__(256) k_cvt(const float* in, _Float16* out, int n) {
  int i = blockIdx.x * blockDim.x + threadIdx.x;
  if (i < n) out[i] = (_Float16)in[i];
}

// W2t[n][k] = W2[k][n] (64x64), f32 -> f16
__global__ void __launch_bounds__(256) k_transpose64(const float* in, _Float16* out) {
  int i = blockIdx.x * blockDim.x + threadIdx.x;
  if (i >= 64 * 64) return;
  int n = i >> 6, k = i & 63;
  out[n * 64 + k] = (_Float16)in[k * 64 + n];
}

__global__ void __launch_bounds__(256) k_addvec(const float* a, const float* b, float* o, int n) {
  int i = blockIdx.x * blockDim.x + threadIdx.x;
  if (i < n) o[i] = a[i] + b[i];
}

// ------------------------------------------------------- graph normalization
__global__ void __launch_bounds__(256) k_deg(const int* dst, const float* ew, float* deg) {
  int i = blockIdx.x * blockDim.x + threadIdx.x;
  if (i >= EPN) return;
  if (i < EE) atomicAdd(&deg[dst[i]], ew[i]);
  else        atomicAdd(&deg[i - EE], 1.f);
}

__global__ void __launch_bounds__(256) k_dis(const float* deg, float* dis) {
  int i = blockIdx.x * blockDim.x + threadIdx.x;
  if (i >= NN) return;
  float d = deg[i];
  dis[i] = (d > 0.f) ? rsqrtf(fmaxf(d, 1e-12f)) : 0.f;
}

__global__ void __launch_bounds__(256) k_norm(const int* src, const int* dst, const float* ew,
                                              const float* dis, float* nrm) {
  int i = blockIdx.x * blockDim.x + threadIdx.x;
  if (i >= EPN) return;
  int s = (i < EE) ? src[i] : i - EE;
  int d = (i < EE) ? dst[i] : i - EE;
  float w = (i < EE) ? ew[i] : 1.f;
  nrm[i] = dis[s] * w * dis[d];
}

// ------------------------------------------------------------- GCN layer 1
// h(bt,n,:) = Xp @ W1 ; X layout (B,N,NF,T): X[(b*NN+n)*64 + f*8 + t]
__global__ void __launch_bounds__(256) k_lin1(const float* X, const float* W1, float* h) {
  long long idx = (long long)blockIdx.x * blockDim.x + threadIdx.x;
  if (idx >= (long long)GROWS * 64) return;
  int c = (int)(idx & 63);
  long long row = idx >> 6;
  int n = (int)(row % NN);
  int bt = (int)(row / NN);
  int b = bt >> 3, t = bt & 7;
  const float* xp = X + ((long long)(b * NN + n)) * 64 + t;  // f stride = T = 8
  float acc = 0.f;
#pragma unroll
  for (int f = 0; f < 8; ++f) acc += xp[f * 8] * W1[f * 64 + c];
  h[idx] = acc;
}

// ---------------------------------------------------------------- scatter
// out[bt, dst, :] += norm * h[bt, src, :], thread = (edge, bt, 8-ch chunk)
__global__ void __launch_bounds__(256) k_scatter(const int* src, const int* dst,
                                                 const float* nrm, const float* h, float* out) {
  long long tid = (long long)blockIdx.x * blockDim.x + threadIdx.x;
  if (tid >= (long long)EPN * 16 * 8) return;
  int c8 = (int)(tid & 7);
  int bt = (int)((tid >> 3) & 15);
  int i  = (int)(tid >> 7);
  int s = (i < EE) ? src[i] : i - EE;
  int d = (i < EE) ? dst[i] : i - EE;
  float w = nrm[i];
  const float4* hp = (const float4*)(h + ((long long)bt * NN + s) * 64 + c8 * 8);
  float*        op = out + ((long long)bt * NN + d) * 64 + c8 * 8;
  float4 a = hp[0], b4 = hp[1];
  atomicAdd(op + 0, w * a.x);
  atomicAdd(op + 1, w * a.y);
  atomicAdd(op + 2, w * a.z);
  atomicAdd(op + 3, w * a.w);
  atomicAdd(op + 4, w * b4.x);
  atomicAdd(op + 5, w * b4.y);
  atomicAdd(op + 6, w * b4.z);
  atomicAdd(op + 7, w * b4.w);
}

// ----------------------------------------------- ReLU(+bias) + BN statistics
// block = 256 threads handles 256 rows; channel = tid & 63
__global__ void __launch_bounds__(256) k_relu_stats(float* x, const float* bias, float* stats) {
  __shared__ float ss[256], sq[256];
  const int tid = threadIdx.x;
  const int c = tid & 63;
  const float b = bias[c];
  float s = 0.f, q = 0.f;
  int rbase = blockIdx.x * 256;
#pragma unroll 4
  for (int it = 0; it < 64; ++it) {
    long long row = rbase + it * 4 + (tid >> 6);
    long long idx = row * 64 + c;
    float v = fmaxf(x[idx] + b, 0.f);
    x[idx] = v;
    s += v;
    q += v * v;
  }
  ss[tid] = s; sq[tid] = q;
  __syncthreads();
  if (tid < 64) {
    s = ss[tid] + ss[tid + 64] + ss[tid + 128] + ss[tid + 192];
    q = sq[tid] + sq[tid + 64] + sq[tid + 128] + sq[tid + 192];
    atomicAdd(&stats[c], s);
    atomicAdd(&stats[64 + c], q);
  }
}

// ------------------------------------------------- BN apply, pack to f16
__global__ void __launch_bounds__(256) k_bn_pack(const float* x, const float* stats,
                                                 const float* g, const float* be, _Float16* out) {
  long long idx = (long long)blockIdx.x * blockDim.x + threadIdx.x;
  if (idx >= (long long)GROWS * 64) return;
  int c = (int)(idx & 63);
  const float inv = 1.f / (float)GROWS;
  float m = stats[c] * inv;
  float v = stats[64 + c] * inv - m * m;
  float r = rsqrtf(v + 1e-5f);
  out[idx] = (_Float16)(g[c] * (x[idx] - m) * r + be[c]);
}

// ------------------------------------------------- GCN2 linear: WMMA GEMM
// C(160000x64) = A(160000x64,f16) @ W2(64x64); W2t is out-major (n-major).
__global__ void __launch_bounds__(256) k_gemm_h(const _Float16* A, const _Float16* W2t, float* C) {
  const int wave = threadIdx.x >> 5;
  const int lane = threadIdx.x & 31;
  const long long r0 = ((long long)blockIdx.x * 8 + wave) * 16;
  v8f acc[4];
#pragma unroll
  for (int nt = 0; nt < 4; ++nt) acc[nt] = zero8();
#pragma unroll
  for (int kc = 0; kc < 2; ++kc) {
    v16h a = frag_rows(A + r0 * 64 + kc * 32, 64);
#pragma unroll
    for (int nt = 0; nt < 4; ++nt) {
      v16h b = frag_rows(W2t + (nt * 16) * 64 + kc * 32, 64);
      acc[nt] = __builtin_amdgcn_wmma_f32_16x16x32_f16(false, a, false, b, (short)0,
                                                       acc[nt], false, false);
    }
  }
#pragma unroll
  for (int nt = 0; nt < 4; ++nt)
#pragma unroll
    for (int i = 0; i < 8; ++i) {
      int m = i + ((lane >> 4) << 3);
      C[(r0 + m) * 64 + nt * 16 + (lane & 15)] = acc[nt][i];
    }
}

// --------------------------------------------------- skip features (f16)
// skip[b*N+n][t*8+f] = X[(b*N+n)*64 + f*8 + t]
__global__ void __launch_bounds__(256) k_skip_pack(const float* X, _Float16* skp) {
  long long idx = (long long)blockIdx.x * blockDim.x + threadIdx.x;
  if (idx >= (long long)BNR * 64) return;
  int c = (int)(idx & 63);
  long long r = idx >> 6;
  int t = c >> 3, f = c & 7;
  skp[idx] = (_Float16)X[r * 64 + f * 8 + t];
}

// ------------------------------------ fused LSTM1 + LSTM2 + FC1 + FC2 (WMMA)
// block = 128 threads (4 waves) owns 16 rows. Wave w computes gate columns
// [w*64, w*64+64) = gates i/f/g/o. h/c state lives in LDS across T steps.
// All LSTM weight fragments (40 x v16h = 320 VGPRs) are pinned in registers
// for the whole T loop -- wave32 on CDNA5 has up to 1024 VGPRs per wave.
__global__ void __launch_bounds__(128) k_lstm_fc(
    const _Float16* x1, const _Float16* x2, const _Float16* skp,
    const _Float16* Wih1, const _Float16* Whh1, const float* bs1,
    const _Float16* Wih2, const _Float16* Whh2, const float* bs2,
    const _Float16* Wfc1, const float* bfc1, const float* Wfc2, const float* bfc2,
    float* out) {
  __shared__ __align__(16) float    gpre[16][256];
  __shared__ __align__(16) _Float16 h1s[16][64];
  __shared__ __align__(16) _Float16 h2s[16][64];
  __shared__ __align__(16) float    c1s[16][64];
  __shared__ __align__(16) float    c2s[16][64];
  __shared__ __align__(16) float    fcs[16][64];

  const int tid = threadIdx.x;
  const int wave = tid >> 5;
  const int lane = tid & 31;
  const long long r0 = (long long)blockIdx.x * 16;
  const int b = (int)(r0 / NN);
  const int n0 = (int)(r0 - (long long)b * NN);  // 16-row tiles never cross b (NN%16==0)

  for (int m = tid; m < 16 * 64; m += 128) {
    ((float*)c1s)[m] = 0.f;
    ((float*)c2s)[m] = 0.f;
    ((_Float16*)h1s)[m] = (_Float16)0.f;
    ((_Float16*)h2s)[m] = (_Float16)0.f;
  }

  // ---- pin all weight fragments in VGPRs (L2-resident, loaded once) ----
  v16h wf1[6][4];
#pragma unroll
  for (int kc = 0; kc < 6; ++kc)
#pragma unroll
    for (int nt = 0; nt < 4; ++nt) {
      int ncol = wave * 64 + nt * 16;
      const _Float16* wb = (kc < 4) ? (Wih1 + (long long)ncol * 128 + kc * 32)
                                    : (Whh1 + (long long)ncol * 64 + (kc - 4) * 32);
      wf1[kc][nt] = frag_rows(wb, (kc < 4) ? 128 : 64);
    }
  v16h wf2[4][4];
#pragma unroll
  for (int kc = 0; kc < 4; ++kc)
#pragma unroll
    for (int nt = 0; nt < 4; ++nt) {
      int ncol = wave * 64 + nt * 16;
      const _Float16* wb = (kc < 2) ? (Wih2 + (long long)ncol * 64 + kc * 32)
                                    : (Whh2 + (long long)ncol * 64 + (kc - 2) * 32);
      wf2[kc][nt] = frag_rows(wb, 64);
    }
  __syncthreads();

  const int arow = tid >> 3;        // cell-update row for this thread
  const int ajb  = (tid & 7) * 8;   // first hidden channel

  for (int t = 0; t < TT; ++t) {
    const _Float16* xa = x1 + ((long long)(b * TT + t) * NN + n0) * 64;
    const _Float16* xb = x2 + ((long long)(b * TT + t) * NN + n0) * 64;

    // prefetch next step's activation tiles (global_prefetch_b8)
    if (t + 1 < TT) {
      __builtin_prefetch(xa + (long long)NN * 64 + (lane & 15) * 64, 0, 3);
      __builtin_prefetch(xb + (long long)NN * 64 + (lane & 15) * 64, 0, 3);
    }

    // ----- LSTM1 gates: K = 128 (x) + 64 (h1)
    v8f acc[4];
#pragma unroll
    for (int nt = 0; nt < 4; ++nt) acc[nt] = zero8();
#pragma unroll
    for (int kc = 0; kc < 6; ++kc) {
      v16h a;
      if (kc < 2)      a = frag_rows(xa + kc * 32, 64);
      else if (kc < 4) a = frag_rows(xb + (kc - 2) * 32, 64);
      else             a = frag_rows(&h1s[0][0] + (kc - 4) * 32, 64);
#pragma unroll
      for (int nt = 0; nt < 4; ++nt)
        acc[nt] = __builtin_amdgcn_wmma_f32_16x16x32_f16(false, a, false, wf1[kc][nt], (short)0,
                                                         acc[nt], false, false);
    }
#pragma unroll
    for (int nt = 0; nt < 4; ++nt)
#pragma unroll
      for (int i = 0; i < 8; ++i) {
        int m = i + ((lane >> 4) << 3);
        int col = wave * 64 + nt * 16 + (lane & 15);
        gpre[m][col] = acc[nt][i] + bs1[col];
      }
    __syncthreads();
#pragma unroll
    for (int u = 0; u < 8; ++u) {
      int j = ajb + u;
      float gi = sigm(gpre[arow][j]);
      float gf = sigm(gpre[arow][64 + j]);
      float gg = tanhf(gpre[arow][128 + j]);
      float go = sigm(gpre[arow][192 + j]);
      float c = gf * c1s[arow][j] + gi * gg;
      c1s[arow][j] = c;
      h1s[arow][j] = (_Float16)(go * tanhf(c));
    }
    __syncthreads();

    // ----- LSTM2 gates: K = 64 (h1) + 64 (h2)
    v8f acc2[4];
#pragma unroll
    for (int nt = 0; nt < 4; ++nt) acc2[nt] = zero8();
#pragma unroll
    for (int kc = 0; kc < 4; ++kc) {
      v16h a = (kc < 2) ? frag_rows(&h1s[0][0] + kc * 32, 64)
                        : frag_rows(&h2s[0][0] + (kc - 2) * 32, 64);
#pragma unroll
      for (int nt = 0; nt < 4; ++nt)
        acc2[nt] = __builtin_amdgcn_wmma_f32_16x16x32_f16(false, a, false, wf2[kc][nt], (short)0,
                                                          acc2[nt], false, false);
    }
#pragma unroll
    for (int nt = 0; nt < 4; ++nt)
#pragma unroll
      for (int i = 0; i < 8; ++i) {
        int m = i + ((lane >> 4) << 3);
        int col = wave * 64 + nt * 16 + (lane & 15);
        gpre[m][col] = acc2[nt][i] + bs2[col];
      }
    __syncthreads();
#pragma unroll
    for (int u = 0; u < 8; ++u) {
      int j = ajb + u;
      float gi = sigm(gpre[arow][j]);
      float gf = sigm(gpre[arow][64 + j]);
      float gg = tanhf(gpre[arow][128 + j]);
      float go = sigm(gpre[arow][192 + j]);
      float c = gf * c2s[arow][j] + gi * gg;
      c2s[arow][j] = c;
      h2s[arow][j] = (_Float16)(go * tanhf(c));
    }
    __syncthreads();
  }

  // ----- FC1: relu([hn1|hn2|skip] @ Wfc1^T + bfc1), K = 192, N = 64
  v8f fa = zero8();
#pragma unroll
  for (int kc = 0; kc < 6; ++kc) {
    v16h a;
    if (kc < 2)      a = frag_rows(&h1s[0][0] + kc * 32, 64);
    else if (kc < 4) a = frag_rows(&h2s[0][0] + (kc - 2) * 32, 64);
    else             a = frag_rows(skp + r0 * 64 + (kc - 4) * 32, 64);
    v16h bf = frag_rows(Wfc1 + (long long)(wave * 16) * 192 + kc * 32, 192);
    fa = __builtin_amdgcn_wmma_f32_16x16x32_f16(false, a, false, bf, (short)0, fa, false, false);
  }
#pragma unroll
  for (int i = 0; i < 8; ++i) {
    int m = i + ((lane >> 4) << 3);
    int col = wave * 16 + (lane & 15);
    fcs[m][col] = fmaxf(fa[i] + bfc1[col], 0.f);
  }
  __syncthreads();

  // ----- FC2: (1 x 64) dot per row
  if (tid < 16) {
    float s = bfc2[0];
#pragma unroll
    for (int k = 0; k < 64; ++k) s += fcs[tid][k] * Wfc2[k];
    out[r0 + tid] = fmaxf(s, 0.f);
  }
}

// ============================================================ host launcher
extern "C" void kernel_launch(void* const* d_in, const int* in_sizes, int n_in,
                              void* d_out, int out_size, void* d_ws, size_t ws_size,
                              hipStream_t stream) {
  const float* X    = (const float*)d_in[0];
  const int*   ei   = (const int*)d_in[1];
  const float* ew   = (const float*)d_in[2];
  const float* W1   = (const float*)d_in[3];
  const float* b1   = (const float*)d_in[4];
  const float* W2   = (const float*)d_in[5];
  const float* b2   = (const float*)d_in[6];
  const float* g1   = (const float*)d_in[7];
  const float* be1  = (const float*)d_in[8];
  const float* g2   = (const float*)d_in[9];
  const float* be2  = (const float*)d_in[10];
  const float* Wih1 = (const float*)d_in[11];
  const float* Whh1 = (const float*)d_in[12];
  const float* bih1 = (const float*)d_in[13];
  const float* bhh1 = (const float*)d_in[14];
  const float* Wih2 = (const float*)d_in[15];
  const float* Whh2 = (const float*)d_in[16];
  const float* bih2 = (const float*)d_in[17];
  const float* bhh2 = (const float*)d_in[18];
  const float* Wfc1 = (const float*)d_in[19];
  const float* bfc1 = (const float*)d_in[20];
  const float* Wfc2 = (const float*)d_in[21];
  const float* bfc2 = (const float*)d_in[22];
  const int* srcI = ei;
  const int* dstI = ei + EE;

  char* w = (char*)d_ws;
  auto take = [&](size_t bytes) -> char* {
    char* p = w;
    w += (bytes + 255) & ~(size_t)255;
    return p;
  };
  float* deg      = (float*)take((size_t)NN * 4);
  float* dis      = (float*)take((size_t)NN * 4);
  float* nrm      = (float*)take((size_t)EPN * 4);
  float* stats    = (float*)take(128 * 4);
  float* hbuf     = (float*)take((size_t)GROWS * 64 * 4);
  float* agg      = (float*)take((size_t)GROWS * 64 * 4);
  _Float16* x1h   = (_Float16*)take((size_t)GROWS * 64 * 2);
  _Float16* x2h   = (_Float16*)take((size_t)GROWS * 64 * 2);
  _Float16* skh   = (_Float16*)take((size_t)BNR * 64 * 2);
  _Float16* W2th  = (_Float16*)take(64 * 64 * 2);
  _Float16* Wih1h = (_Float16*)take(256 * 128 * 2);
  _Float16* Whh1h = (_Float16*)take(256 * 64 * 2);
  _Float16* Wih2h = (_Float16*)take(256 * 64 * 2);
  _Float16* Whh2h = (_Float16*)take(256 * 64 * 2);
  _Float16* Wfc1h = (_Float16*)take(64 * 192 * 2);
  float* bs1      = (float*)take(256 * 4);
  float* bs2      = (float*)take(256 * 4);

  auto blocks = [](long long n, int bs) { return dim3((unsigned)((n + bs - 1) / bs)); };

  // weight prep
  k_transpose64<<<blocks(64 * 64, 256), 256, 0, stream>>>(W2, W2th);
  k_cvt<<<blocks(256 * 128, 256), 256, 0, stream>>>(Wih1, Wih1h, 256 * 128);
  k_cvt<<<blocks(256 * 64, 256), 256, 0, stream>>>(Whh1, Whh1h, 256 * 64);
  k_cvt<<<blocks(256 * 64, 256), 256, 0, stream>>>(Wih2, Wih2h, 256 * 64);
  k_cvt<<<blocks(256 * 64, 256), 256, 0, stream>>>(Whh2, Whh2h, 256 * 64);
  k_cvt<<<blocks(64 * 192, 256), 256, 0, stream>>>(Wfc1, Wfc1h, 64 * 192);
  k_addvec<<<1, 256, 0, stream>>>(bih1, bhh1, bs1, 256);
  k_addvec<<<1, 256, 0, stream>>>(bih2, bhh2, bs2, 256);
  k_skip_pack<<<blocks((long long)BNR * 64, 256), 256, 0, stream>>>(X, skh);

  // graph normalization
  k_zero<<<blocks(NN, 256), 256, 0, stream>>>(deg, NN);
  k_deg<<<blocks(EPN, 256), 256, 0, stream>>>(dstI, ew, deg);
  k_dis<<<blocks(NN, 256), 256, 0, stream>>>(deg, dis);
  k_norm<<<blocks(EPN, 256), 256, 0, stream>>>(srcI, dstI, ew, dis, nrm);

  const long long NELT = (long long)GROWS * 64;
  const long long SJOBS = (long long)EPN * 16 * 8;

  // ----- GCN layer 1
  k_lin1<<<blocks(NELT, 256), 256, 0, stream>>>(X, W1, hbuf);
  k_zero<<<blocks(NELT, 256), 256, 0, stream>>>(agg, NELT);
  k_scatter<<<blocks(SJOBS, 256), 256, 0, stream>>>(srcI, dstI, nrm, hbuf, agg);
  k_zero<<<1, 128, 0, stream>>>(stats, 128);
  k_relu_stats<<<GROWS / 256, 256, 0, stream>>>(agg, b1, stats);
  k_bn_pack<<<blocks(NELT, 256), 256, 0, stream>>>(agg, stats, g1, be1, x1h);

  // ----- GCN layer 2 (WMMA GEMM)
  k_gemm_h<<<GROWS / 128, 256, 0, stream>>>(x1h, W2th, hbuf);
  k_zero<<<blocks(NELT, 256), 256, 0, stream>>>(agg, NELT);
  k_scatter<<<blocks(SJOBS, 256), 256, 0, stream>>>(srcI, dstI, nrm, hbuf, agg);
  k_zero<<<1, 128, 0, stream>>>(stats, 128);
  k_relu_stats<<<GROWS / 256, 256, 0, stream>>>(agg, b2, stats);
  k_bn_pack<<<blocks(NELT, 256), 256, 0, stream>>>(agg, stats, g2, be2, x2h);

  // ----- fused LSTM1+LSTM2+FC1+FC2
  k_lstm_fc<<<BNR / 16, 128, 0, stream>>>(x1h, x2h, skh,
                                          Wih1h, Whh1h, bs1,
                                          Wih2h, Whh2h, bs2,
                                          Wfc1h, bfc1, Wfc2, bfc2,
                                          (float*)d_out);
}
